// DKVMN_5334349382444
// MI455X (gfx1250) — compile-verified, matrix operand order checked by reading, not run on hardware
//
#include <hip/hip_runtime.h>
#include <hip/hip_bf16.h>
#include <math.h>

// Sizes from the reference
#define BQ   2048
#define SEQ  200
#define KD   64     // key / query dim
#define VD   128    // value dim
#define CD   256    // memory slots
#define SUMD 128
#define CATD 192    // VD + KD

typedef _Float16 v16h __attribute__((ext_vector_type(16)));
typedef float    v8f  __attribute__((ext_vector_type(8)));

// ---- pre-swizzled B-fragment table in d_ws ----
// fragment = 16x32 (KxN=32x16) f16 tile in WMMA B layout; per lane 16 halves stored
// contiguously (32B) -> one vector load per fragment in the main kernel.
#define KT_KM 2            // KD/32 k-steps for key_mem GEMMs
#define NT_KM 16           // CD/16 n-tiles
#define OFF_KM 0
#define KT_WE 4            // VD/32
#define NT_WE 8            // VD/16
#define OFF_WE (OFF_KM + KT_KM * NT_KM)      // 32
#define OFF_WA (OFF_WE + KT_WE * NT_WE)      // 64
#define KT_VM 8            // CD/32
#define NT_VM 8            // VD/16
#define OFF_VM (OFF_WA + KT_WE * NT_WE)      // 96
#define KT_WS 6            // CATD/32
#define NT_WS 8            // SUMD/16
#define OFF_WS (OFF_VM + KT_VM * NT_VM)      // 160
#define NFRAG  (OFF_WS + KT_WS * NT_WS)      // 208 fragments * 1KB = 208KB of d_ws

// CDNA5 16-bit A/B fragment K mapping (CDNA5 ISA §7.12.2):
// lanes 0-15: K = {0..7, 16..23}; lanes 16-31: K = {8..15, 24..31}; 2 halves/VGPR.
__device__ __forceinline__ int klocal(int e, int hi) {
  return (e & 7) + (hi ? 8 : 0) + ((e >= 8) ? 16 : 0);
}

// A fragment: 16 x 32 tile, row m = lane&15, from f16 LDS row-major [16 x stride].
// Each 8-element half is contiguous -> lowers to 2x ds_load_b128.
__device__ __forceinline__ v16h load_a_h(const _Float16* src, int stride, int kbase, int lane) {
  const int hi = (lane >> 4) & 1, m = lane & 15;
  v16h a;
#pragma unroll
  for (int e = 0; e < 16; ++e) a[e] = src[m * stride + kbase + klocal(e, hi)];
  return a;
}

// Pre-swizzled B fragment: one 32B vector load per lane.
__device__ __forceinline__ v16h load_frag(const _Float16* wsbuf, int f, int lane) {
  return ((const v16h*)wsbuf)[f * 32 + lane];
}

__device__ __forceinline__ v8f wmma16(v16h a, v16h b, v8f c) {
  return __builtin_amdgcn_wmma_f32_16x16x32_f16(false, a, false, b, (short)0, c, false, false);
}

// -------- prep: convert + swizzle shared weights into WMMA B-fragment layout --------
__global__ __launch_bounds__(32)
void prep_frags_kernel(const float* __restrict__ key_mem, const float* __restrict__ We,
                       const float* __restrict__ Wa, const float* __restrict__ value_mem,
                       const float* __restrict__ Ws, _Float16* __restrict__ wsbuf)
{
  const int f = blockIdx.x;
  const int lane = threadIdx.x;
  const int hi = (lane >> 4) & 1, l15 = lane & 15;
  v16h frag;

  if (f < OFF_WE) {                       // key_mem: row-major KxN, B[k][n] = KM[k*CD + n]
    const int fi = f - OFF_KM, kt = fi / NT_KM, nt = fi % NT_KM;
    const int kb = kt * 32, n = nt * 16 + l15;
#pragma unroll
    for (int e = 0; e < 16; ++e) frag[e] = (_Float16)key_mem[(kb + klocal(e, hi)) * CD + n];
  } else if (f < OFF_WA) {                // We.T: B[k][n] = We[n*VD + k]
    const int fi = f - OFF_WE, kt = fi / NT_WE, nt = fi % NT_WE;
    const int kb = kt * 32, n = nt * 16 + l15;
#pragma unroll
    for (int e = 0; e < 16; ++e) frag[e] = (_Float16)We[n * VD + kb + klocal(e, hi)];
  } else if (f < OFF_VM) {                // Wa.T
    const int fi = f - OFF_WA, kt = fi / NT_WE, nt = fi % NT_WE;
    const int kb = kt * 32, n = nt * 16 + l15;
#pragma unroll
    for (int e = 0; e < 16; ++e) frag[e] = (_Float16)Wa[n * VD + kb + klocal(e, hi)];
  } else if (f < OFF_WS) {                // value_mem.T: B[k=c][n=v] = vm[v*CD + c]
    const int fi = f - OFF_VM, kt = fi / NT_VM, nt = fi % NT_VM;
    const int kb = kt * 32, n = nt * 16 + l15;
#pragma unroll
    for (int e = 0; e < 16; ++e) frag[e] = (_Float16)value_mem[n * CD + kb + klocal(e, hi)];
  } else {                                // Ws.T: B[k][n] = Ws[n*CATD + k]
    const int fi = f - OFF_WS, kt = fi / NT_WS, nt = fi % NT_WS;
    const int kb = kt * 32, n = nt * 16 + l15;
#pragma unroll
    for (int e = 0; e < 16; ++e) frag[e] = (_Float16)Ws[n * CATD + kb + klocal(e, hi)];
  }
  ((v16h*)wsbuf)[f * 32 + lane] = frag;
}

// -------- fused main kernel: one wave handles 16 batch rows end-to-end --------
__global__ __launch_bounds__(32)
void dkvmn_fused_kernel(const int* __restrict__ input_ids, const int* __restrict__ tag_id,
                        const int* __restrict__ target_id,
                        const float* __restrict__ q_emb, const float* __restrict__ i_emb,
                        const float* __restrict__ be, const float* __restrict__ ba,
                        const float* __restrict__ bs, const float* __restrict__ Wo,
                        const float* __restrict__ bo, const _Float16* __restrict__ wsbuf,
                        float* __restrict__ out)
{
  __shared__ _Float16 s_qvt[16 * KD];    // q_emb[tag_id[:, -1]]      (A for w logits)
  __shared__ _Float16 s_qvq[16 * KD];    // q_emb[target_id]          (A for wt logits + concat tail)
  __shared__ _Float16 s_iv [16 * VD];    // i_emb[input_ids[:, -1]]   (A for e/a)
  __shared__ _Float16 s_w  [16 * CD];    // w logits -> softmax w -> p = w*wt (f16)
  __shared__ _Float16 s_wt [16 * CD];    // wt logits -> softmax wt (f16)
  __shared__ float    s_e  [16 * VD];    // sigmoid gate; later reused as summary
  __shared__ float    s_a  [16 * VD];    // tanh add
  __shared__ _Float16 s_x  [16 * CATD];  // concat([read, qv])
  __shared__ float    s_s  [16];         // s = sum_c w*wt per row
  __shared__ int      s_qT [16];
  __shared__ int      s_iT [16];
  __shared__ int      s_tq [16];

  const int lane = threadIdx.x;          // wave32
  const int b0   = blockIdx.x * 16;
  const int hi   = (lane >> 4) & 1;
  const int l15  = lane & 15;

  if (lane < 16) {
    const int b = b0 + lane;
    s_qT[lane] = tag_id   [b * SEQ + SEQ - 1];
    s_iT[lane] = input_ids[b * SEQ + SEQ - 1];
    s_tq[lane] = target_id[b];
  }
  __syncthreads();

  // Gather the three embedding rows per batch row -> f16 LDS
  for (int i = lane; i < 16 * KD; i += 32) {
    const int r = i >> 6, c = i & (KD - 1);
    s_qvt[i] = (_Float16)q_emb[s_qT[r] * KD + c];
    s_qvq[i] = (_Float16)q_emb[s_tq[r] * KD + c];
  }
  for (int i = lane; i < 16 * VD; i += 32) {
    const int r = i >> 7, c = i & (VD - 1);
    s_iv[i] = (_Float16)i_emb[s_iT[r] * VD + c];
  }
  __syncthreads();

  // ---- logits: (16xK) @ key_mem (KxC), shared B fragments for w & wt ----
  for (int nt = 0; nt < NT_KM; ++nt) {
    v8f c0 = {}, c1 = {};
#pragma unroll
    for (int kt = 0; kt < KT_KM; ++kt) {
      const v16h bf = load_frag(wsbuf, OFF_KM + kt * NT_KM + nt, lane);
      c0 = wmma16(load_a_h(s_qvt, KD, kt * 32, lane), bf, c0);
      c1 = wmma16(load_a_h(s_qvq, KD, kt * 32, lane), bf, c1);
    }
    const int n = nt * 16 + l15;
#pragma unroll
    for (int r = 0; r < 8; ++r) {
      const int m = r + hi * 8;
      s_w [m * CD + n] = (_Float16)c0[r];
      s_wt[m * CD + n] = (_Float16)c1[r];
    }
  }

  // ---- e/a: iv (16x128) @ We.T / Wa.T, fused bias + activation ----
  for (int nt = 0; nt < NT_WE; ++nt) {
    v8f ce = {}, ca = {};
#pragma unroll
    for (int kt = 0; kt < KT_WE; ++kt) {
      const v16h af = load_a_h(s_iv, VD, kt * 32, lane);
      ce = wmma16(af, load_frag(wsbuf, OFF_WE + kt * NT_WE + nt, lane), ce);
      ca = wmma16(af, load_frag(wsbuf, OFF_WA + kt * NT_WE + nt, lane), ca);
    }
    const int n = nt * 16 + l15;
    const float bev = be[n], bav = ba[n];
#pragma unroll
    for (int r = 0; r < 8; ++r) {
      const int m = r + hi * 8;
      s_e[m * VD + n] = 1.0f / (1.0f + __expf(-(ce[r] + bev)));
      s_a[m * VD + n] = tanhf(ca[r] + bav);
    }
  }
  __syncthreads();

  // ---- softmax rows of s_w and s_wt (f32 math, f16 storage); p = w*wt -> s_w; s = rowsum(p) ----
  for (int m = 0; m < 16; ++m) {
#pragma unroll
    for (int which = 0; which < 2; ++which) {
      _Float16* row = which ? (s_wt + m * CD) : (s_w + m * CD);
      float v[CD / 32];
      float mx = -3.0e38f;
#pragma unroll
      for (int j = 0; j < CD / 32; ++j) { v[j] = (float)row[lane + j * 32]; mx = fmaxf(mx, v[j]); }
#pragma unroll
      for (int off = 16; off > 0; off >>= 1) mx = fmaxf(mx, __shfl_xor(mx, off));
      float sum = 0.f;
#pragma unroll
      for (int j = 0; j < CD / 32; ++j) { v[j] = __expf(v[j] - mx); sum += v[j]; }
#pragma unroll
      for (int off = 16; off > 0; off >>= 1) sum += __shfl_xor(sum, off);
      const float inv = 1.0f / sum;
#pragma unroll
      for (int j = 0; j < CD / 32; ++j) row[lane + j * 32] = (_Float16)(v[j] * inv);
    }
    float ps = 0.f;
#pragma unroll
    for (int j = 0; j < CD / 32; ++j) {
      const int c = lane + j * 32;
      const float p = (float)s_w[m * CD + c] * (float)s_wt[m * CD + c];
      s_w[m * CD + c] = (_Float16)p;
      ps += p;
    }
#pragma unroll
    for (int off = 16; off > 0; off >>= 1) ps += __shfl_xor(ps, off);
    if (lane == 0) s_s[m] = ps;
  }
  __syncthreads();

  // ---- read = wt@vm^T - e*(p@vm^T) + a*s   (crnt never materialized) ----
  for (int nt = 0; nt < NT_VM; ++nt) {
    v8f c0 = {}, c1 = {};
#pragma unroll
    for (int kt = 0; kt < KT_VM; ++kt) {
      const v16h bv = load_frag(wsbuf, OFF_VM + kt * NT_VM + nt, lane);
      c0 = wmma16(load_a_h(s_wt, CD, kt * 32, lane), bv, c0);
      c1 = wmma16(load_a_h(s_w,  CD, kt * 32, lane), bv, c1);
    }
    const int n = nt * 16 + l15;  // value index v
#pragma unroll
    for (int r = 0; r < 8; ++r) {
      const int m = r + hi * 8;
      const float rd = c0[r] - s_e[m * VD + n] * c1[r] + s_a[m * VD + n] * s_s[m];
      s_x[m * CATD + n] = (_Float16)rd;
    }
  }
  // concat tail: x[:, 128:192] = qv(target)
  for (int i = lane; i < 16 * KD; i += 32) {
    const int r = i >> 6, c = i & (KD - 1);
    s_x[r * CATD + VD + c] = s_qvq[r * KD + c];
  }
  __syncthreads();

  // ---- summary = tanh(x @ Ws.T + bs)  (reuse s_e as summary storage) ----
  for (int nt = 0; nt < NT_WS; ++nt) {
    v8f c = {};
#pragma unroll
    for (int kt = 0; kt < KT_WS; ++kt) {
      c = wmma16(load_a_h(s_x, CATD, kt * 32, lane),
                 load_frag(wsbuf, OFF_WS + kt * NT_WS + nt, lane), c);
    }
    const int n = nt * 16 + l15;
    const float bsv = bs[n];
#pragma unroll
    for (int r = 0; r < 8; ++r) {
      const int m = r + hi * 8;
      s_e[m * SUMD + n] = tanhf(c[r] + bsv);
    }
  }
  __syncthreads();

  // ---- out = summary @ Wo.T + bo ----
  const float bo0 = bo[0];
  for (int m = 0; m < 16; ++m) {
    float acc = 0.f;
#pragma unroll
    for (int j = 0; j < SUMD / 32; ++j) {
      const int k = lane + j * 32;
      acc += s_e[m * SUMD + k] * Wo[k];
    }
#pragma unroll
    for (int off = 16; off > 0; off >>= 1) acc += __shfl_xor(acc, off);
    if (lane == 0) out[b0 + m] = acc + bo0;
  }
}

extern "C" void kernel_launch(void* const* d_in, const int* in_sizes, int n_in,
                              void* d_out, int out_size, void* d_ws, size_t ws_size,
                              hipStream_t stream) {
  const int*   input_ids = (const int*)  d_in[0];
  const int*   tag_id    = (const int*)  d_in[1];
  const int*   target_id = (const int*)  d_in[2];
  const float* q_emb     = (const float*)d_in[3];
  const float* i_emb     = (const float*)d_in[4];
  const float* key_mem   = (const float*)d_in[5];
  const float* value_mem = (const float*)d_in[6];
  const float* We        = (const float*)d_in[7];
  const float* be        = (const float*)d_in[8];
  const float* Wa        = (const float*)d_in[9];
  const float* ba        = (const float*)d_in[10];
  const float* Ws        = (const float*)d_in[11];
  const float* bs        = (const float*)d_in[12];
  const float* Wo        = (const float*)d_in[13];
  const float* bo        = (const float*)d_in[14];
  float* out = (float*)d_out;
  _Float16* wsbuf = (_Float16*)d_ws;   // needs NFRAG*32*16*2 = 208KB of scratch

  // 1) one-time (per launch) weight swizzle into WMMA B-fragment layout
  hipLaunchKernelGGL(prep_frags_kernel, dim3(NFRAG), dim3(32), 0, stream,
                     key_mem, We, Wa, value_mem, Ws, wsbuf);
  // 2) fused DKVMN pipeline, 16 batch rows per wave (stream order guarantees prep done)
  hipLaunchKernelGGL(dkvmn_fused_kernel, dim3(BQ / 16), dim3(32), 0, stream,
                     input_ids, tag_id, target_id, q_emb, i_emb,
                     be, ba, bs, Wo, bo, wsbuf, out);
}
